// MLP_fusion_70085276336620
// MI455X (gfx1250) — matrix operand order repaired and checked by reading, not compiled, not run on hardware
//
#include <hip/hip_runtime.h>

typedef __attribute__((ext_vector_type(16))) __bf16 v16bf;
typedef __attribute__((ext_vector_type(8)))  __bf16 v8bf;
typedef __attribute__((ext_vector_type(8)))  float  v8f;
typedef __attribute__((ext_vector_type(4)))  unsigned int v4u;
typedef __attribute__((ext_vector_type(8)))  int v8i;
typedef __attribute__((ext_vector_type(4)))  int v4i;

#define TPB 256
#define ROWS 64
#define PITCH_H 520   // 512 + 8 pad, bf16 elems (row stride 1040B = 4 banks mod 64)
#define PITCH_S 136   // 128 + 8 pad
#define ACT_BYTES (2*ROWS*PITCH_H*2 + 2*ROWS*PITCH_S*2 + ROWS*128*4)  // 200704
#define STG_BYTES (16*1024)           // 16 weight tiles per staged chunk
#define STG0_OFF  ACT_BYTES
#define STG1_OFF  (ACT_BYTES + STG_BYTES)
#define SMEM_BYTES (ACT_BYTES + 2*STG_BYTES)   // 233472

// ---- packed-weight offsets in d_ws (bf16 elements) ----
enum {
  OFF_SUBW1 = 0,
  OFF_SUBW2 = OFF_SUBW1 + 128*512,
  OFF_SUBWD = OFF_SUBW2 + 512*512,
  OFF_MMW1  = OFF_SUBWD + 512*64,
  OFF_MMW2  = OFF_MMW1  + 128*512,
  OFF_MMWD  = OFF_MMW2  + 512*512,
  OFF_ATTW  = OFF_MMWD  + 512*64,
  OFF_FUSW1 = OFF_ATTW  + 128*128,
  OFF_FUSW2 = OFF_FUSW1 + 128*512,
  OFF_FUSWD = OFF_FUSW2 + 512*512,
  OFF_END   = OFF_FUSWD + 512*64
};

// Repack fp32 W[K][N] into bf16 WMMA B-fragment tiles (32x16 per tile).
// Tile t = nt*(K/32)+kt, element (lane l, idx i) = W[kt*32 + (i/8)*16 + (l/16)*8 + i%8][nt*16 + (l%16)]
__global__ void repack_w_kernel(const float* __restrict__ W, __bf16* __restrict__ out,
                                int K, int N) {
  int KT = K >> 5;
  int total = K * N;
  for (int o = blockIdx.x * blockDim.x + threadIdx.x; o < total;
       o += gridDim.x * blockDim.x) {
    int i = o & 15;
    int l = (o >> 4) & 31;
    int tile = o >> 9;
    int kt = tile % KT;
    int nt = tile / KT;
    int k = kt * 32 + (i >> 3) * 16 + (l >> 4) * 8 + (i & 7);
    int n = nt * 16 + (l & 15);
    out[o] = (__bf16)W[k * N + n];
  }
}

// TDM: stage `ntiles` packed 1KB weight tiles (row stride = strideUnits*8 bytes)
// from global into LDS at byte offset ldsOff. 2-D descriptor, data_size = 8B.
__device__ __forceinline__ void tdm_stage(const __bf16* gsrc, unsigned ldsOff,
                                          unsigned strideUnits, unsigned ntiles) {
  unsigned long long ga = (unsigned long long)(size_t)gsrc;
  v4u g0;
  g0.x = 1u;                                            // count=1 (valid), no gather
  g0.y = ldsOff;                                        // lds_addr (bytes)
  g0.z = (unsigned)ga;                                  // global_addr[31:0]
  g0.w = (unsigned)((ga >> 32) & 0x1FFFFFFu) | (2u << 30); // addr[56:32] | type=2
  v8i g1;
  g1[0] = (int)(3u << 16);                              // data_size = 8 bytes
  g1[1] = (int)((strideUnits & 0xFFFFu) << 16);         // tensor_dim0[15:0]
  g1[2] = (int)(((strideUnits >> 16) & 0xFFFFu)         // tensor_dim0[31:16]
                | ((ntiles & 0xFFFFu) << 16));          // tensor_dim1[15:0]
  g1[3] = (int)(128u << 16);                            // tile_dim0 = 128 units (1KB)
  g1[4] = (int)ntiles;                                  // tile_dim1, tile_dim2=0
  g1[5] = (int)strideUnits;                             // tensor_dim0_stride[31:0]
  g1[6] = 0;                                            // stride[47:32], dim1_stride lo
  g1[7] = 0;
  v4i z4 = {0, 0, 0, 0};
  v8i z8 = {0, 0, 0, 0, 0, 0, 0, 0};
  __builtin_amdgcn_tensor_load_to_lds(g0, g1, z4, z4, z8, 0);
}

// A fragment: 16x32 bf16, per documented layout. lane m=lane%16, h=lane/16:
// elems 0..7 = K(kBase + h*8 .. +7), elems 8..15 = K(kBase + 16 + h*8 .. +7)
__device__ __forceinline__ v16bf load_a_frag(const __bf16* act, int pitch,
                                             int rowBase, int kBase) {
  int lane = threadIdx.x & 31;
  int m = lane & 15, h = lane >> 4;
  const __bf16* p0 = act + (rowBase + m) * pitch + kBase + h * 8;
  v8bf lo = *(const v8bf*)p0;
  v8bf hi = *(const v8bf*)(p0 + 16);
  return __builtin_shufflevector(lo, hi, 0,1,2,3,4,5,6,7,8,9,10,11,12,13,14,15);
}

// MODE 0: relu -> bf16 LDS out
// MODE 1: none -> f32 sCat[:,catCol+..] + bf16 sC16 + f32 global out
// MODE 2: sigmoid(acc+b) * sCat -> bf16 LDS out (attention gate)
// MODE 3: none -> f32 global out only
template<int K, int N, int MODE>
__device__ __forceinline__ void gemm(const __bf16* actIn, int inPitch,
                                     const __bf16* wPacked, const float* bias,
                                     __bf16* sStg0, __bf16* sStg1,
                                     __bf16* outBf, int outPitch,
                                     float* catF, __bf16* catBf, int catCol,
                                     float* gOut) {
  constexpr int KT = K / 32;
  constexpr int NTILES = N / 16;
  constexpr int GT = (NTILES >= 16) ? 16 : NTILES;  // column tiles per group
  constexpr int NGROUPS = NTILES / GT;
  constexpr int PW = GT / 2;                        // tiles per wave (2 waves/row-tile)

  const int wid  = threadIdx.x >> 5;
  const int lane = threadIdx.x & 31;
  const int rt   = wid & 3;
  const int half = wid >> 2;
  const int n    = lane & 15;
  const int h    = lane >> 4;
  const int rowBase = rt * 16;

  for (int g = 0; g < NGROUPS; ++g) {
    if (wid == 0)   // prologue: stage chunk 0 of this group
      tdm_stage(wPacked + (size_t)(g * GT) * KT * 512, STG0_OFF, KT * 128, GT);

    v8f acc[PW];
    #pragma unroll
    for (int j = 0; j < PW; ++j)
      acc[j] = (v8f){0.f,0.f,0.f,0.f,0.f,0.f,0.f,0.f};

    for (int kc = 0; kc < KT; ++kc) {
      if (wid == 0) __builtin_amdgcn_s_wait_tensorcnt(0);
      __syncthreads();   // chunk kc resident; previous reads of next buffer done
      if (wid == 0 && kc + 1 < KT)
        tdm_stage(wPacked + ((size_t)(g * GT) * KT + (kc + 1)) * 512,
                  (kc & 1) ? STG0_OFF : STG1_OFF, KT * 128, GT);

      const __bf16* sb = (kc & 1) ? sStg1 : sStg0;
      v16bf a = load_a_frag(actIn, inPitch, rowBase, kc * 32);
      #pragma unroll
      for (int j = 0; j < PW; ++j) {
        v16bf b = *(const v16bf*)(sb + (half * PW + j) * 512 + lane * 16);
        acc[j] = __builtin_amdgcn_wmma_f32_16x16x32_bf16(
            false, a, false, b, (short)0, acc[j], false, false);
      }
    }

    #pragma unroll
    for (int j = 0; j < PW; ++j) {
      const int col = (g * GT + half * PW + j) * 16 + n;
      const float bv = bias[col];
      #pragma unroll
      for (int r = 0; r < 8; ++r) {
        const int row = rowBase + r + 8 * h;   // C/D layout: M = r + 8*(lane/16)
        float v = acc[j][r] + bv;
        if (MODE == 0) {
          v = v > 0.f ? v : 0.f;
          outBf[row * outPitch + col] = (__bf16)v;
        } else if (MODE == 1) {
          catF[row * 128 + catCol + col] = v;
          catBf[row * PITCH_S + catCol + col] = (__bf16)v;
          gOut[row * 64 + col] = v;
        } else if (MODE == 2) {
          float s = 1.f / (1.f + __expf(-v));
          float cv = catF[row * 128 + col];
          outBf[row * outPitch + col] = (__bf16)(s * cv);
        } else { // MODE 3
          gOut[row * 64 + col] = v;
        }
      }
    }
  }
}

__global__ __launch_bounds__(TPB, 1)
void fused_beam_kernel(const float* __restrict__ x, const float* __restrict__ y,
                       const __bf16* __restrict__ wp,
                       const float* sub_b1, const float* sub_b2, const float* sub_bd,
                       const float* mm_b1,  const float* mm_b2,  const float* mm_bd,
                       const float* att_b,
                       const float* fus_b1, const float* fus_b2, const float* fus_bd,
                       float* __restrict__ out, int B) {
  extern __shared__ char smem[];
  __bf16* sH   = (__bf16*)smem;               // [64][520]
  __bf16* sG   = sH  + ROWS * PITCH_H;        // [64][520]
  __bf16* sIn  = sG  + ROWS * PITCH_H;        // [64][136]
  __bf16* sC16 = sIn + ROWS * PITCH_S;        // [64][136] cat as bf16
  float*  sCat = (float*)(sC16 + ROWS * PITCH_S); // [64][128] cat as f32
  __bf16* sStg0 = (__bf16*)(smem + STG0_OFF); // 16KB weight stage ping
  __bf16* sStg1 = (__bf16*)(smem + STG1_OFF); // 16KB weight stage pong

  const int row0 = blockIdx.x * ROWS;
  const size_t B64 = (size_t)B * 64;

  // ---- load x tile -> bf16 LDS ----
  for (int e = threadIdx.x; e < ROWS * 128; e += TPB) {
    int row = e >> 7, col = e & 127;
    sIn[row * PITCH_S + col] = (__bf16)x[(size_t)(row0 + row) * 128 + col];
  }
  __syncthreads();

  // ---- sub MLP ----
  gemm<128,512,0>(sIn, PITCH_S, wp+OFF_SUBW1, sub_b1, sStg0,sStg1, sH, PITCH_H,
                  nullptr,nullptr,0,nullptr);
  __syncthreads();
  gemm<512,512,0>(sH, PITCH_H, wp+OFF_SUBW2, sub_b2, sStg0,sStg1, sG, PITCH_H,
                  nullptr,nullptr,0,nullptr);
  __syncthreads();
  gemm<512,64,1>(sG, PITCH_H, wp+OFF_SUBWD, sub_bd, sStg0,sStg1, nullptr,0,
                 sCat, sC16, 0, out + (size_t)row0 * 64);
  __syncthreads();

  // ---- top-8 of out_sub (wb_num==1 -> p_mm == out_sub), build masked y ----
  if (threadIdx.x < ROWS) {
    const int row = threadIdx.x;
    unsigned* zp = (unsigned*)(sIn + row * PITCH_S);
    #pragma unroll
    for (int c = 0; c < 64; ++c) zp[c] = 0u;   // zero 128 bf16 cols
    const float* yrow = y + (size_t)(row0 + row) * 128;
    unsigned long long selmask = 0ull;
    for (int s = 0; s < 8; ++s) {
      float best = -3.4e38f; int bi = 0;
      for (int j = 0; j < 64; ++j) {
        float v = sCat[row * 128 + j];
        if (!((selmask >> j) & 1ull) && v > best) { best = v; bi = j; }
      }
      selmask |= (1ull << bi);
      sIn[row * PITCH_S + 2 * bi]     = (__bf16)yrow[2 * bi];
      sIn[row * PITCH_S + 2 * bi + 1] = (__bf16)yrow[2 * bi + 1];
    }
  }
  __syncthreads();

  // ---- mm MLP ----
  gemm<128,512,0>(sIn, PITCH_S, wp+OFF_MMW1, mm_b1, sStg0,sStg1, sH, PITCH_H,
                  nullptr,nullptr,0,nullptr);
  __syncthreads();
  gemm<512,512,0>(sH, PITCH_H, wp+OFF_MMW2, mm_b2, sStg0,sStg1, sG, PITCH_H,
                  nullptr,nullptr,0,nullptr);
  __syncthreads();
  gemm<512,64,1>(sG, PITCH_H, wp+OFF_MMWD, mm_bd, sStg0,sStg1, nullptr,0,
                 sCat, sC16, 64, out + B64 + (size_t)row0 * 64);
  __syncthreads();

  // ---- attention gate: sIn = bf16( sigmoid(cat@W+b) * cat ) ----
  gemm<128,128,2>(sC16, PITCH_S, wp+OFF_ATTW, att_b, sStg0,sStg1, sIn, PITCH_S,
                  sCat, nullptr, 0, nullptr);
  __syncthreads();

  // ---- fusion MLP ----
  gemm<128,512,0>(sIn, PITCH_S, wp+OFF_FUSW1, fus_b1, sStg0,sStg1, sH, PITCH_H,
                  nullptr,nullptr,0,nullptr);
  __syncthreads();
  gemm<512,512,0>(sH, PITCH_H, wp+OFF_FUSW2, fus_b2, sStg0,sStg1, sG, PITCH_H,
                  nullptr,nullptr,0,nullptr);
  __syncthreads();
  gemm<512,64,3>(sG, PITCH_H, wp+OFF_FUSWD, fus_bd, sStg0,sStg1, nullptr,0,
                 nullptr,nullptr,0, out + 2 * B64 + (size_t)row0 * 64);
}

extern "C" void kernel_launch(void* const* d_in, const int* in_sizes, int n_in,
                              void* d_out, int out_size, void* d_ws, size_t ws_size,
                              hipStream_t stream) {
  (void)n_in; (void)out_size; (void)ws_size;
  const float* x      = (const float*)d_in[0];
  const float* y      = (const float*)d_in[1];
  const float* sub_w1 = (const float*)d_in[2];  const float* sub_b1 = (const float*)d_in[3];
  const float* sub_w2 = (const float*)d_in[4];  const float* sub_b2 = (const float*)d_in[5];
  const float* sub_wd = (const float*)d_in[6];  const float* sub_bd = (const float*)d_in[7];
  const float* mm_w1  = (const float*)d_in[8];  const float* mm_b1  = (const float*)d_in[9];
  const float* mm_w2  = (const float*)d_in[10]; const float* mm_b2  = (const float*)d_in[11];
  const float* mm_wd  = (const float*)d_in[12]; const float* mm_bd  = (const float*)d_in[13];
  const float* att_w  = (const float*)d_in[14]; const float* att_b  = (const float*)d_in[15];
  const float* fus_w1 = (const float*)d_in[16]; const float* fus_b1 = (const float*)d_in[17];
  const float* fus_w2 = (const float*)d_in[18]; const float* fus_b2 = (const float*)d_in[19];
  const float* fus_wd = (const float*)d_in[20]; const float* fus_bd = (const float*)d_in[21];
  float* out = (float*)d_out;
  __bf16* wp = (__bf16*)d_ws;
  const int B = in_sizes[0] / 128;

  auto rp = [&](const float* W, int K, int N, int off) {
    int total = K * N;
    repack_w_kernel<<<(total + TPB - 1) / TPB, TPB, 0, stream>>>(W, wp + off, K, N);
  };
  rp(sub_w1, 128, 512, OFF_SUBW1);
  rp(sub_w2, 512, 512, OFF_SUBW2);
  rp(sub_wd, 512,  64, OFF_SUBWD);
  rp(mm_w1,  128, 512, OFF_MMW1);
  rp(mm_w2,  512, 512, OFF_MMW2);
  rp(mm_wd,  512,  64, OFF_MMWD);
  rp(att_w,  128, 128, OFF_ATTW);
  rp(fus_w1, 128, 512, OFF_FUSW1);
  rp(fus_w2, 512, 512, OFF_FUSW2);
  rp(fus_wd, 512,  64, OFF_FUSWD);

  (void)hipFuncSetAttribute((const void*)fused_beam_kernel,
                            hipFuncAttributeMaxDynamicSharedMemorySize, SMEM_BYTES);
  fused_beam_kernel<<<B / ROWS, TPB, SMEM_BYTES, stream>>>(
      x, y, wp,
      sub_b1, sub_b2, sub_bd,
      mm_b1,  mm_b2,  mm_bd,
      att_b,
      fus_b1, fus_b2, fus_bd,
      out, B);
}
